// VQVAE_5282809774153
// MI455X (gfx1250) — matrix operand order repaired and checked by reading, not compiled, not run on hardware
//
#include <hip/hip_runtime.h>
#include <hip/hip_bf16.h>

typedef __attribute__((ext_vector_type(16))) _Float16 v16h;
typedef __attribute__((ext_vector_type(8)))  float    v8f;

#define AP  36   // As row stride: [m][k], 32 k + pad   (72 B rows)
#define BP  40   // Bs row stride: [n][k], 32 k + pad   (80 B rows)
#define VBP 72   // VQ tiles row stride: [n][d]/[m][d], 64 d + pad

// ---------------------------------------------------------------------------
// Fused im2col + WMMA conv / deconv, templated on geometry so all k-decode
// divisions fold to shifts/constant magic. Ping-pong LDS double buffering:
// one barrier per K chunk; staging of chunk i overlaps WMMAs of chunk i-1.
// Block: 256 thr = 8 waves. Block tile: 32 (Cout) x 128 (N = B*Hout*Wout).
// Wave (wm 0..1, wn 0..3): 16 x 32 -> two v_wmma_f32_16x16x32_f16 per chunk.
// Normal conv : py = oy*STRIDE - PAD + ky              (LDIL == 1)
// Deconv      : py = oy - PAD' + ky, py even, py >>= 1 (LDIL == 2, STRIDE == 1)
// ---------------------------------------------------------------------------
template<int KH, int KW, int STRIDE, int PAD, int LDIL>
__global__ void __launch_bounds__(256) conv_wmma_kernel(
    const float* __restrict__ in, const _Float16* __restrict__ wt,
    const float* __restrict__ bias, float* __restrict__ out,
    int Cin, int Hin, int Win,
    int Cout, int Hout, int Wout,
    int K, int Kpad, int Ntot, int act)
{
    __shared__ _Float16 As[2][32 * AP];     // [m][k]
    __shared__ _Float16 Bs[2][128 * BP];    // [n][k]  (K-contiguous rows)

    constexpr int KHW = KH * KW;

    const int tid  = threadIdx.x;
    const int lane = tid & 31;
    const int wave = tid >> 5;
    const int wm   = wave >> 2;          // 0..1 : 16-row Cout subtile
    const int wn   = wave & 3;           // 0..3 : 32-col N subtile
    const int HWo  = Hout * Wout;
    const int n0   = blockIdx.x * 128;
    const int m0   = blockIdx.y * 32;

    // ---- per-thread B staging decode: column index is K-chunk invariant ----
    const int snn = tid & 127;           // staged n within tile (fixed)
    const int skb = tid >> 7;            // k parity (0/1), covers k via +2*i
    const int sgn = n0 + snn;
    const bool svalid = sgn < Ntot;
    int soy = 0, sox = 0;
    const float* sbase = in;
    if (svalid) {
        int b = sgn / HWo;
        int s = sgn - b * HWo;
        soy = s / Wout;  sox = s - soy * Wout;
        soy = soy * STRIDE - PAD;
        sox = sox * STRIDE - PAD;
        sbase = in + (size_t)b * Cin * Hin * Win;
    }
    // ---- per-thread A staging decode ----
    const int akk = tid & 31;
    const int amb = tid >> 5;

    const int nchunks = Kpad >> 5;
    v8f acc0 = {}, acc1 = {};

    for (int ic = 0; ic < nchunks; ++ic) {
        const int k0 = ic << 5;
        const int p  = ic & 1;
        // ---- stage A tile (weights 32x32); prefetch next chunk ----
        #pragma unroll
        for (int i = 0; i < 4; ++i) {
            int m  = amb + 8 * i;
            int gm = m0 + m;
            _Float16 v = (_Float16)0.f;
            if (gm < Cout) v = wt[(size_t)gm * Kpad + k0 + akk];
            As[p][m * AP + akk] = v;
        }
        if (k0 + 32 < Kpad && m0 + amb < Cout)
            __builtin_prefetch(&wt[(size_t)(m0 + amb) * Kpad + k0 + 32 + akk], 0, 1);
        // ---- stage B tile (im2col gather, 128 n x 32 k) ----
        #pragma unroll
        for (int i = 0; i < 16; ++i) {
            const int kk = skb + 2 * i;
            const int gk = k0 + kk;
            float v = 0.f;
            if (svalid && gk < K) {
                int ci = gk / KHW;              // constant divisor -> cheap
                int r  = gk - ci * KHW;
                int ky = r / KW;
                int kx = r - ky * KW;
                int py = soy + ky;
                int px = sox + kx;
                bool ok = (py >= 0) && (px >= 0);
                if (LDIL == 2) { ok = ok && !((py | px) & 1); py >>= 1; px >>= 1; }
                ok = ok && (py < Hin) && (px < Win);
                if (ok) v = sbase[((size_t)ci * Hin + py) * Win + px];
            }
            Bs[p][snn * BP + kk] = (_Float16)v;
        }
        __syncthreads();   // single barrier/chunk; prev compute fenced by prior barrier

        // ---- fragments (contiguous k = 16h + e in A and B), 2 MMAs per wave ----
        v16h a, b0, b1;
        {
            const int mrow = (wm << 4) + (lane & 15);
            const int kb   = (lane < 16) ? 0 : 16;
            #pragma unroll
            for (int e = 0; e < 8; ++e) {
                a[e]     = As[p][mrow * AP + kb + e];
                a[e + 8] = As[p][mrow * AP + kb + 8 + e];
            }
            const int kb2 = (lane < 16) ? 0 : 16;
            const int nr0 = (wn << 5) + (lane & 15);
            #pragma unroll
            for (int e = 0; e < 16; ++e) {
                b0[e] = Bs[p][nr0 * BP + kb2 + e];
                b1[e] = Bs[p][(nr0 + 16) * BP + kb2 + e];
            }
        }
        acc0 = __builtin_amdgcn_wmma_f32_16x16x32_f16(false, a, false, b0, (short)0, acc0, false, false);
        acc1 = __builtin_amdgcn_wmma_f32_16x16x32_f16(false, a, false, b1, (short)0, acc1, false, false);
    }

    // ---- epilogue: bias + activation + NCHW scatter ----
    const int mhalf = (lane < 16) ? 0 : 8;
    #pragma unroll
    for (int t = 0; t < 2; ++t) {
        const int gn = n0 + (wn << 5) + (t << 4) + (lane & 15);
        if (gn < Ntot) {
            int b = gn / HWo;
            int s = gn - b * HWo;
            #pragma unroll
            for (int r = 0; r < 8; ++r) {
                int gm = m0 + (wm << 4) + mhalf + r;
                if (gm < Cout) {
                    float y = (t == 0) ? acc0[r] : acc1[r];
                    if (bias) y += bias[gm];
                    if (act == 1)      y = (y > 0.f) ? y : 0.01f * y;   // leaky
                    else if (act == 2) y = tanhf(y);
                    out[((size_t)b * Cout + gm) * HWo + s] = y;
                }
            }
        }
    }
}

template<int KH, int KW, int S, int P, int L>
static void launch_conv(const float* in, const _Float16* wt, const float* bias, float* out,
                        int Bn, int Cin, int Hin, int Win, int Cout, int Hout, int Wout,
                        int K, int Kpad, int act, hipStream_t stream)
{
    int Ntot = Bn * Hout * Wout;
    dim3 grid((Ntot + 127) / 128, (Cout + 31) / 32);
    conv_wmma_kernel<KH, KW, S, P, L><<<grid, dim3(256), 0, stream>>>(
        in, wt, bias, out, Cin, Hin, Win, Cout, Hout, Wout, K, Kpad, Ntot, act);
}

// ---------------------------------------------------------------------------
// VQ argmin: dist'(m,n) = |emb_m|^2 - 2 * z_n . emb_m   (|z|^2 const per n)
// 64 z-points per block, 512 codes chunked by 32, WMMA dot products, packed
// (orderable-float | idx) u64 LDS atomicMin for argmin (ties -> lowest idx).
// ---------------------------------------------------------------------------
__global__ void __launch_bounds__(256) vq_argmin_kernel(
    const float* __restrict__ z,          // encoded, NCHW [64,64,32,32]
    const _Float16* __restrict__ emb16,   // [512, 64]
    const float* __restrict__ en2,        // [512]  |emb|^2
    int* __restrict__ idxout)             // [65536]
{
    __shared__ _Float16 Bs[64 * VBP];     // [n][d]  z tile, row-contiguous in d
    __shared__ _Float16 As[32 * VBP];     // [m][d]  code chunk
    __shared__ float sen2[512];
    __shared__ unsigned long long sbest[64];

    const int tid = threadIdx.x, lane = tid & 31, wave = tid >> 5;
    const int wm = wave >> 2, wn = wave & 3;
    const int n0 = blockIdx.x * 64;
    const int HW = 1024, C = 64;

    for (int e = tid; e < 512; e += 256) sen2[e] = en2[e];
    if (tid < 64) sbest[tid] = ~0ull;

    {   // z tile: thread's n fixed (tid & 63), d strided
        const int nn = tid & 63;
        const int db = tid >> 6;
        const int gn = n0 + nn;
        const int b = gn >> 10, s = gn & 1023;
        #pragma unroll
        for (int i = 0; i < 16; ++i) {
            int d = db + 4 * i;
            Bs[nn * VBP + d] = (_Float16)z[((size_t)b * C + d) * HW + s];
        }
    }
    __syncthreads();

    // preload both B fragments (k0 = 0 and 32); reused for all 16 code chunks
    v16h bf0, bf1;
    {
        const int nr  = (wn << 4) + (lane & 15);
        const int kb2 = (lane < 16) ? 0 : 16;
        #pragma unroll
        for (int e = 0; e < 16; ++e) {
            bf0[e] = Bs[nr * VBP + kb2 + e];
            bf1[e] = Bs[nr * VBP + 32 + kb2 + e];
        }
    }

    float bestv = 3.4e38f; int besti = 0;
    const int mrow  = (wm << 4) + (lane & 15);
    const int kb    = (lane < 16) ? 0 : 16;
    const int mhalf = (lane < 16) ? 0 : 8;

    for (int mc = 0; mc < 512; mc += 32) {
        {   // code chunk: thread's d fixed (tid & 63), m strided
            const int d  = tid & 63;
            const int mb = tid >> 6;
            #pragma unroll
            for (int i = 0; i < 8; ++i) {
                int m = mb + 4 * i;
                As[m * VBP + d] = emb16[(size_t)(mc + m) * 64 + d];
            }
        }
        __syncthreads();

        v8f c = {};
        v16h a;
        #pragma unroll
        for (int e = 0; e < 8; ++e) {
            a[e]     = As[mrow * VBP + kb + e];
            a[e + 8] = As[mrow * VBP + kb + 8 + e];
        }
        c = __builtin_amdgcn_wmma_f32_16x16x32_f16(false, a, false, bf0, (short)0, c, false, false);
        #pragma unroll
        for (int e = 0; e < 8; ++e) {
            a[e]     = As[mrow * VBP + 32 + kb + e];
            a[e + 8] = As[mrow * VBP + 32 + kb + 8 + e];
        }
        c = __builtin_amdgcn_wmma_f32_16x16x32_f16(false, a, false, bf1, (short)0, c, false, false);

        #pragma unroll
        for (int r = 0; r < 8; ++r) {
            int code = mc + (wm << 4) + mhalf + r;
            float d2 = sen2[code] - 2.f * c[r];
            if (d2 < bestv) { bestv = d2; besti = code; }
        }
        __syncthreads();
    }

    unsigned int u = __float_as_uint(bestv);
    u = (u & 0x80000000u) ? ~u : (u | 0x80000000u);   // monotone order-preserving
    unsigned long long packed = ((unsigned long long)u << 32) | (unsigned int)besti;
    atomicMin(&sbest[(wn << 4) + (lane & 15)], packed);
    __syncthreads();
    if (tid < 64) idxout[n0 + tid] = (int)(sbest[tid] & 0xFFFFFFFFu);
}

// ---------------------------------------------------------------------------
// Gather q = emb[idx]; write `quantized` (forward value of straight-through
// estimator) and accumulate vq_loss = beta * mean((q - z)^2)
// ---------------------------------------------------------------------------
__global__ void __launch_bounds__(256) vq_gather_kernel(
    const float* __restrict__ z, const float* __restrict__ emb,
    const int* __restrict__ idx, float* __restrict__ q_out,
    float* __restrict__ loss)
{
    __shared__ float red[256];
    int p = blockIdx.x * 256 + threadIdx.x;   // point id 0..65535
    int b = p >> 10, s = p & 1023;
    int id = idx[p];
    float acc = 0.f;
    for (int d = 0; d < 64; ++d) {
        float q = emb[(size_t)id * 64 + d];
        size_t off = ((size_t)b * 64 + d) * 1024 + s;
        float zz = z[off];
        q_out[off] = q;
        float df = q - zz;
        acc += df * df;
    }
    red[threadIdx.x] = acc;
    __syncthreads();
    for (int st = 128; st > 0; st >>= 1) {
        if (threadIdx.x < st) red[threadIdx.x] += red[threadIdx.x + st];
        __syncthreads();
    }
    if (threadIdx.x == 0)
        atomicAdd(loss, red[0] * (0.25f / (65536.f * 64.f)));
}

// ---------------------------------------------------------------------------
// Weight conversion f32 -> f16 [O, Kpad]; deconv flag flips + transposes
// (torch ConvTranspose2d weight [I, O, KH, KW] -> OIHW flipped kernel)
// ---------------------------------------------------------------------------
__global__ void cvt_w_kernel(const float* __restrict__ w, _Float16* __restrict__ o16,
                             int O, int I, int KH, int KW, int Kpad, int deconv)
{
    int i = blockIdx.x * 256 + threadIdx.x;
    if (i >= O * Kpad) return;
    int o = i / Kpad, k = i - o * Kpad;
    int K = I * KH * KW;
    float v = 0.f;
    if (k < K) {
        int khw = KH * KW;
        int ci = k / khw, r = k - ci * khw;
        int ky = r / KW, kx = r - ky * KW;
        if (deconv)
            v = w[(((size_t)ci * O + o) * KH + (KH - 1 - ky)) * KW + (KW - 1 - kx)];
        else
            v = w[(((size_t)o * I + ci) * KH + ky) * KW + kx];
    }
    o16[i] = (_Float16)v;
}

__global__ void emb_norm2_kernel(const float* __restrict__ emb, float* __restrict__ en2)
{
    int c = blockIdx.x * 256 + threadIdx.x;
    if (c >= 512) return;
    float s = 0.f;
    for (int d = 0; d < 64; ++d) { float v = emb[(size_t)c * 64 + d]; s += v * v; }
    en2[c] = s;
}

// ---------------------------------------------------------------------------
// Training-mode BatchNorm2d: batch stats over (N,H,W); one block per channel
// ---------------------------------------------------------------------------
__global__ void __launch_bounds__(256) bn_stats_kernel(
    const float* __restrict__ x, float* __restrict__ mean, float* __restrict__ istd,
    int C, int HW, int Bn)
{
    __shared__ float s1[256], s2[256];
    const int c = blockIdx.x;
    float a = 0.f, b = 0.f;
    for (int bb = 0; bb < Bn; ++bb) {
        const float* row = x + ((size_t)bb * C + c) * HW;
        for (int s = threadIdx.x; s < HW; s += 256) {
            float v = row[s];
            a += v; b += v * v;
        }
    }
    s1[threadIdx.x] = a; s2[threadIdx.x] = b;
    __syncthreads();
    for (int st = 128; st > 0; st >>= 1) {
        if (threadIdx.x < st) { s1[threadIdx.x] += s1[threadIdx.x + st];
                                s2[threadIdx.x] += s2[threadIdx.x + st]; }
        __syncthreads();
    }
    if (threadIdx.x == 0) {
        float tot = (float)(Bn * HW);
        float m = s1[0] / tot;
        float v = s2[0] / tot - m * m;
        mean[c] = m;
        istd[c] = rsqrtf(v + 1e-5f);
    }
}

// mode 0: plain   mode 1: + relu   mode 2: + residual skip add
// Geometry fixed to [64, 64, 32, 32] (HW = 1024, C = 64) -> shift/mask decode.
__global__ void bn_apply_kernel(const float* __restrict__ x, const float* __restrict__ skip,
    float* __restrict__ out, const float* __restrict__ mean, const float* __restrict__ istd,
    const float* __restrict__ g, const float* __restrict__ beta,
    int total, int mode)
{
    int i = blockIdx.x * 256 + threadIdx.x;
    if (i >= total) return;
    int c = (i >> 10) & 63;
    float y = (x[i] - mean[c]) * istd[c] * g[c] + beta[c];
    if (mode == 1)      y = fmaxf(y, 0.f);
    else if (mode == 2) y += skip[i];
    out[i] = y;
}

__global__ void leaky_kernel(const float* __restrict__ x, float* __restrict__ out, int total)
{
    int i = blockIdx.x * 256 + threadIdx.x;
    if (i >= total) return;
    float v = x[i];
    out[i] = v > 0.f ? v : 0.01f * v;
}

__global__ void zero1_kernel(float* p)
{
    if (threadIdx.x == 0 && blockIdx.x == 0) *p = 0.f;
}

// ---------------------------------------------------------------------------
extern "C" void kernel_launch(void* const* d_in, const int* in_sizes, int n_in,
                              void* d_out, int out_size, void* d_ws, size_t ws_size,
                              hipStream_t stream)
{
    (void)in_sizes; (void)n_in; (void)out_size; (void)ws_size;

    // inputs (setup_inputs dict order)
    const float* x    = (const float*)d_in[0];
    const float* e1w  = (const float*)d_in[1];  const float* e1b  = (const float*)d_in[2];
    const float* e2w  = (const float*)d_in[3];  const float* e2b  = (const float*)d_in[4];
    const float* e3w  = (const float*)d_in[5];  const float* e3b  = (const float*)d_in[6];
    const float* er1w = (const float*)d_in[7];  const float* er1g = (const float*)d_in[8];
    const float* er1b = (const float*)d_in[9];
    const float* er2w = (const float*)d_in[10]; const float* er2g = (const float*)d_in[11];
    const float* er2b = (const float*)d_in[12];
    const float* e4w  = (const float*)d_in[13]; const float* e4b  = (const float*)d_in[14];
    const float* emb  = (const float*)d_in[15];
    const float* d1w  = (const float*)d_in[16]; const float* d1b  = (const float*)d_in[17];
    const float* dr1w = (const float*)d_in[18]; const float* dr1g = (const float*)d_in[19];
    const float* dr1b = (const float*)d_in[20];
    const float* dr2w = (const float*)d_in[21]; const float* dr2g = (const float*)d_in[22];
    const float* dr2b = (const float*)d_in[23];
    const float* dt1w = (const float*)d_in[24]; const float* dt1b = (const float*)d_in[25];
    const float* dt2w = (const float*)d_in[26]; const float* dt2b = (const float*)d_in[27];

    // outputs concatenated: recon | quantized | encoded | vq_loss
    float* recon     = (float*)d_out;                 // 64*3*128*128  = 3145728
    float* quantized = recon + 3145728;               // 64*64*32*32   = 4194304
    float* encoded   = quantized + 4194304;           // 64*64*32*32   = 4194304
    float* lossp     = encoded + 4194304;             // scalar

    // workspace suballocation
    char*  ws  = (char*)d_ws;
    size_t off = 0;
    auto alloc = [&](size_t bytes) -> void* {
        void* p = ws + off;
        off += (bytes + 255) & ~(size_t)255;
        return p;
    };
    _Float16* w_e1  = (_Float16*)alloc(32  * 64   * 2);
    _Float16* w_e2  = (_Float16*)alloc(64  * 512  * 2);
    _Float16* w_e3  = (_Float16*)alloc(64  * 576  * 2);
    _Float16* w_er1 = (_Float16*)alloc(64  * 576  * 2);
    _Float16* w_er2 = (_Float16*)alloc(64  * 64   * 2);
    _Float16* w_e4  = (_Float16*)alloc(64  * 64   * 2);
    _Float16* w_d1  = (_Float16*)alloc(64  * 576  * 2);
    _Float16* w_dr1 = (_Float16*)alloc(64  * 576  * 2);
    _Float16* w_dr2 = (_Float16*)alloc(64  * 64   * 2);
    _Float16* w_dt1 = (_Float16*)alloc(32  * 1024 * 2);
    _Float16* w_dt2 = (_Float16*)alloc(3   * 512  * 2);
    _Float16* emb16 = (_Float16*)alloc(512 * 64   * 2);
    float* en2b  = (float*)alloc(512 * 4);
    float* meanb = (float*)alloc(64 * 4);
    float* istdb = (float*)alloc(64 * 4);
    int*   idxb  = (int*)alloc(65536 * 4);
    float* actA = (float*)alloc((size_t)8388608 * 4);   // [64,32,64,64]
    float* actB = (float*)alloc((size_t)4194304 * 4);   // [64,64,32,32]
    float* actC = (float*)alloc((size_t)4194304 * 4);
    float* actD = (float*)alloc((size_t)4194304 * 4);

    const dim3 blk(256);
    auto cvt = [&](const float* w, _Float16* o16, int O, int I, int KH, int KW, int Kp, int dec) {
        int tot = O * Kp;
        cvt_w_kernel<<<(tot + 255) / 256, blk, 0, stream>>>(w, o16, O, I, KH, KW, Kp, dec);
    };
    cvt(e1w,  w_e1,  32, 3,  4, 4, 64,   0);
    cvt(e2w,  w_e2,  64, 32, 4, 4, 512,  0);
    cvt(e3w,  w_e3,  64, 64, 3, 3, 576,  0);
    cvt(er1w, w_er1, 64, 64, 3, 3, 576,  0);
    cvt(er2w, w_er2, 64, 64, 1, 1, 64,   0);
    cvt(e4w,  w_e4,  64, 64, 1, 1, 64,   0);
    cvt(d1w,  w_d1,  64, 64, 3, 3, 576,  0);
    cvt(dr1w, w_dr1, 64, 64, 3, 3, 576,  0);
    cvt(dr2w, w_dr2, 64, 64, 1, 1, 64,   0);
    cvt(dt1w, w_dt1, 32, 64, 4, 4, 1024, 1);
    cvt(dt2w, w_dt2, 3,  32, 4, 4, 512,  1);
    cvt(emb,  emb16, 512, 64, 1, 1, 64,  0);
    emb_norm2_kernel<<<2, blk, 0, stream>>>(emb, en2b);

    // ---------------- encoder ----------------
    launch_conv<4,4,2,1,1>(x,    w_e1, e1b, actA, 64, 3,  128, 128, 32, 64, 64, 48,  64,  1, stream);
    launch_conv<4,4,2,1,1>(actA, w_e2, e2b, actB, 64, 32, 64,  64,  64, 32, 32, 512, 512, 1, stream);
    launch_conv<3,3,1,1,1>(actB, w_e3, e3b, actC, 64, 64, 32,  32,  64, 32, 32, 576, 576, 1, stream);

    // resblock(actC): conv3x3 -> bn -> relu -> conv1x1 -> bn -> +skip
    launch_conv<3,3,1,1,1>(actC, w_er1, nullptr, actD, 64, 64, 32, 32, 64, 32, 32, 576, 576, 0, stream);
    bn_stats_kernel<<<64, blk, 0, stream>>>(actD, meanb, istdb, 64, 1024, 64);
    bn_apply_kernel<<<16384, blk, 0, stream>>>(actD, nullptr, actD, meanb, istdb,
                                               er1g, er1b, 4194304, 1);
    launch_conv<1,1,1,0,1>(actD, w_er2, nullptr, actB, 64, 64, 32, 32, 64, 32, 32, 64, 64, 0, stream);
    bn_stats_kernel<<<64, blk, 0, stream>>>(actB, meanb, istdb, 64, 1024, 64);
    bn_apply_kernel<<<16384, blk, 0, stream>>>(actB, actC, actB, meanb, istdb,
                                               er2g, er2b, 4194304, 2);
    leaky_kernel<<<16384, blk, 0, stream>>>(actB, actB, 4194304);

    launch_conv<1,1,1,0,1>(actB, w_e4, e4b, encoded, 64, 64, 32, 32, 64, 32, 32, 64, 64, 1, stream);

    // ---------------- vector quantize ----------------
    vq_argmin_kernel<<<1024, blk, 0, stream>>>(encoded, emb16, en2b, idxb);
    zero1_kernel<<<1, blk, 0, stream>>>(lossp);
    vq_gather_kernel<<<256, blk, 0, stream>>>(encoded, emb, idxb, quantized, lossp);

    // ---------------- decoder ----------------
    launch_conv<3,3,1,1,1>(quantized, w_d1, d1b, actC, 64, 64, 32, 32, 64, 32, 32, 576, 576, 1, stream);

    launch_conv<3,3,1,1,1>(actC, w_dr1, nullptr, actD, 64, 64, 32, 32, 64, 32, 32, 576, 576, 0, stream);
    bn_stats_kernel<<<64, blk, 0, stream>>>(actD, meanb, istdb, 64, 1024, 64);
    bn_apply_kernel<<<16384, blk, 0, stream>>>(actD, nullptr, actD, meanb, istdb,
                                               dr1g, dr1b, 4194304, 1);
    launch_conv<1,1,1,0,1>(actD, w_dr2, nullptr, actB, 64, 64, 32, 32, 64, 32, 32, 64, 64, 0, stream);
    bn_stats_kernel<<<64, blk, 0, stream>>>(actB, meanb, istdb, 64, 1024, 64);
    bn_apply_kernel<<<16384, blk, 0, stream>>>(actB, actC, actB, meanb, istdb,
                                               dr2g, dr2b, 4194304, 2);
    leaky_kernel<<<16384, blk, 0, stream>>>(actB, actB, 4194304);

    // deconvs: lhs-dilated (LDIL=2) conv with flipped kernel, pad' = 4-1-1 = 2
    launch_conv<4,4,1,2,2>(actB, w_dt1, dt1b, actA, 64, 64, 32, 32, 32, 64, 64,  1024, 1024, 1, stream);
    launch_conv<4,4,1,2,2>(actA, w_dt2, dt2b, recon, 64, 32, 64, 64, 3, 128, 128, 512,  512,  2, stream);
}
